// sLSTMCell_45475113730292
// MI455X (gfx1250) — compile-verified, matrix operand order checked by reading, not run on hardware
//
#include <hip/hip_runtime.h>
#include <hip/hip_bf16.h>

#define B_   32768
#define D_   1024
#define NH_  8
#define H_   128
#define K_   4
#define EPS_ 1e-6f
#define CAP_ 15.0f

typedef __bf16 bf16;
typedef __bf16 v16bf __attribute__((ext_vector_type(16)));
typedef __bf16 v8bf  __attribute__((ext_vector_type(8)));
typedef __bf16 v4bf  __attribute__((ext_vector_type(4)));
typedef float  v8f   __attribute__((ext_vector_type(8)));
typedef float  v4f   __attribute__((ext_vector_type(4)));
typedef int    v4i   __attribute__((vector_size(16)));

// ---------------------------------------------------------------------------
// CDNA5 async global->LDS copy (ASYNCcnt path), with plain load/ds-store
// fallback if the builtin is not declared by this toolchain.
// The builtin takes generic vector-typed pointers (v4i*) per hipcc diagnostics.
// ---------------------------------------------------------------------------
#if defined(__gfx1250__) && \
    __has_builtin(__builtin_amdgcn_global_load_async_to_lds_b128) && \
    __has_builtin(__builtin_amdgcn_s_wait_asynccnt)
#define USE_ASYNC_LDS 1
#else
#define USE_ASYNC_LDS 0
#endif

__device__ __forceinline__ void async_cp16(const bf16* g, bf16* l) {
#if USE_ASYNC_LDS
  __builtin_amdgcn_global_load_async_to_lds_b128((v4i*)g, (v4i*)l, 0, 0);
#else
  *(v8bf*)l = *(const v8bf*)g;
#endif
}

__device__ __forceinline__ void async_fence() {
#if USE_ASYNC_LDS
  __builtin_amdgcn_s_wait_asynccnt(0);
#endif
}

// ---------------------------------------------------------------------------
// Conversion kernels
// ---------------------------------------------------------------------------
__global__ __launch_bounds__(256)
void cvt_f32_to_bf16_vec4(const float* __restrict__ src, bf16* __restrict__ dst, int n4) {
  int i = blockIdx.x * blockDim.x + threadIdx.x;
  if (i >= n4) return;
  v4f x = *(const v4f*)(src + 4 * (size_t)i);
  v4bf y;
  y[0] = (bf16)x[0]; y[1] = (bf16)x[1]; y[2] = (bf16)x[2]; y[3] = (bf16)x[3];
  *(v4bf*)(dst + 4 * (size_t)i) = y;
}

// conv_w is [D, D, K]; last tap: dst[o*D + d] = w[(o*D + d)*K + (K-1)]
__global__ __launch_bounds__(256)
void cvt_convw_last(const float* __restrict__ w, bf16* __restrict__ dst, int n) {
  int i = blockIdx.x * blockDim.x + threadIdx.x;
  if (i >= n) return;
  dst[i] = (bf16)w[(size_t)i * K_ + (K_ - 1)];
}

// ---------------------------------------------------------------------------
// bf16 WMMA GEMM: C[M,1024] = A[M,K] * Bw[1024,K]^T
// Block tile 128x256, 8 waves (2 in M x 4 in N), wave tile 64x64, K-step 32.
// 16 WMMAs per 16 ds_read_b128 per wave per K-step.
// MODE 0: SiLU(acc + bias) -> bf16    (x_conv)
// MODE 1: acc -> bf16                 (z)
// MODE 2: acc -> f32                  (out)
// ---------------------------------------------------------------------------
#define TM  128
#define TN  256
#define TK  32
#define LDT 40   // padded row stride (elements); 80B rows keep 16B alignment

template <int MODE>
__global__ __launch_bounds__(256)
void gemm_bf16_wmma(const bf16* __restrict__ A, const bf16* __restrict__ Bw,
                    const float* __restrict__ bias,
                    float* __restrict__ outF, bf16* __restrict__ outBf, int Kdim) {
  __shared__ __align__(16) bf16 Asm[2][TM][LDT];
  __shared__ __align__(16) bf16 Bsm[2][TN][LDT];

  const int tid  = threadIdx.x;
  const int bm   = blockIdx.y * TM;
  const int bn   = blockIdx.x * TN;
  const int wave = tid >> 5;
  const int lane = tid & 31;
  const int lr   = lane & 15;
  const int kh   = (lane < 16) ? 0 : 8;    // per-lane K half per 16-bit A/B layout
  const int wm   = (wave & 1) * 64;        // wave M offset within block tile
  const int wn   = (wave >> 1) * 64;       // wave N offset within block tile
  const int half8 = (lane < 16) ? 0 : 8;   // C/D layout M offset

  v8f acc[4][4];
  const v8f vzero = {0.f, 0.f, 0.f, 0.f, 0.f, 0.f, 0.f, 0.f};
#pragma unroll
  for (int i = 0; i < 4; ++i)
#pragma unroll
    for (int j = 0; j < 4; ++j) acc[i][j] = vzero;

  // Stage A tile (128x32: 512 8-elt chunks) and B tile (256x32: 1024 chunks).
  auto load_tile = [&](int s, int kk) {
#pragma unroll
    for (int u = 0; u < 2; ++u) {
      int ch = tid + u * 256;   // 0..511
      int r  = ch >> 2;
      int c  = (ch & 3) * 8;
      async_cp16(A + (size_t)(bm + r) * Kdim + kk + c, &Asm[s][r][c]);
    }
#pragma unroll
    for (int u = 0; u < 4; ++u) {
      int ch = tid + u * 256;   // 0..1023
      int r  = ch >> 2;
      int c  = (ch & 3) * 8;
      async_cp16(Bw + (size_t)(bn + r) * Kdim + kk + c, &Bsm[s][r][c]);
    }
  };

  union frag { v16bf v; v8bf h[2]; };

  load_tile(0, 0);
  async_fence();
  __syncthreads();

  for (int kk = 0; kk < Kdim; kk += TK) {
    const int s = (kk / TK) & 1;
    if (kk + TK < Kdim) load_tile(s ^ 1, kk + TK);

    // Load all fragments into distinct registers first, then burst WMMAs.
    frag bfr[4], af[4];
#pragma unroll
    for (int j = 0; j < 4; ++j) {
      const bf16* br = &Bsm[s][wn + j * 16 + lr][0];
      bfr[j].h[0] = *(const v8bf*)(br + kh);
      bfr[j].h[1] = *(const v8bf*)(br + kh + 16);
    }
#pragma unroll
    for (int i = 0; i < 4; ++i) {
      const bf16* ar = &Asm[s][wm + i * 16 + lr][0];
      af[i].h[0] = *(const v8bf*)(ar + kh);
      af[i].h[1] = *(const v8bf*)(ar + kh + 16);
    }
#pragma unroll
    for (int i = 0; i < 4; ++i)
#pragma unroll
      for (int j = 0; j < 4; ++j)
        acc[i][j] = __builtin_amdgcn_wmma_f32_16x16x32_bf16(
            false, af[i].v, false, bfr[j].v, (short)0, acc[i][j], false, false);

    async_fence();
    __syncthreads();
  }

  // Epilogue. C/D layout: lanes 0-15 -> M=r, N=lane; lanes 16-31 -> M=8+r, N=lane-16.
#pragma unroll
  for (int i = 0; i < 4; ++i) {
    const int mbase = bm + wm + i * 16 + half8;
#pragma unroll
    for (int j = 0; j < 4; ++j) {
      const int ncol = bn + wn + j * 16 + lr;
#pragma unroll
      for (int r = 0; r < 8; ++r) {
        float v = acc[i][j][r];
        size_t idx = (size_t)(mbase + r) * D_ + ncol;
        if (MODE == 0) {
          float xv  = v + bias[ncol];
          float sig = 1.f / (1.f + __expf(-xv));
          outBf[idx] = (bf16)(xv * sig);
        } else if (MODE == 1) {
          outBf[idx] = (bf16)v;
        } else {
          outF[idx] = v;
        }
      }
    }
  }
}

// ---------------------------------------------------------------------------
// Gating + small projections + per-head layernorm.
// One wave32 per (batch, head); 8 waves per block -> one block per batch row.
// ---------------------------------------------------------------------------
__global__ __launch_bounds__(256)
void gate_ln_kernel(const float* __restrict__ x,     // inputs [B,D]
                    const bf16* __restrict__ xconv,  // [B,D] bf16
                    const bf16* __restrict__ zbf,    // [B,D] bf16
                    const float* __restrict__ c, const float* __restrict__ n,
                    const float* __restrict__ m,
                    const float* __restrict__ iw, const float* __restrict__ ib,
                    const float* __restrict__ fw, const float* __restrict__ fb,
                    const float* __restrict__ ow, const float* __restrict__ ob,
                    const float* __restrict__ gnw, const float* __restrict__ gnb,
                    float* __restrict__ c_out, float* __restrict__ n_out,
                    float* __restrict__ m_out, bf16* __restrict__ hn) {
  const int tid  = threadIdx.x;
  const int wave = tid >> 5;
  const int lane = tid & 31;
  const int p = blockIdx.x * 8 + wave;
  const int b = p >> 3;
  const int h = p & 7;

  const size_t rowx = (size_t)b * D_;
  const size_t roww = (size_t)h * D_;

  // i/f projections on x_conv, o projection on raw inputs (length-1024 dots)
  float si = 0.f, sf = 0.f, so = 0.f;
  for (int k = lane; k < D_; k += 32) {
    float xc = (float)xconv[rowx + k];
    si += xc * iw[roww + k];
    sf += xc * fw[roww + k];
    so += x[rowx + k] * ow[roww + k];
  }
#pragma unroll
  for (int off = 16; off > 0; off >>= 1) {
    si += __shfl_xor(si, off, 32);
    sf += __shfl_xor(sf, off, 32);
    so += __shfl_xor(so, off, 32);
  }

  const float it   = CAP_ * tanhf((si + ib[h]) * (1.f / CAP_));
  const float ft   = CAP_ * tanhf((sf + fb[h]) * (1.f / CAP_));
  const float ot   = CAP_ * tanhf((so + ob[h]) * (1.f / CAP_));
  const float flog = -log1pf(__expf(-ft));            // log(sigmoid(f))
  const float mo   = m[(size_t)b * NH_ + h];
  const float mn   = fmaxf(flog + mo, it);
  const float ig   = __expf(it - mn);
  const float fg   = __expf(flog + mo - mn);
  const float og   = 1.f / (1.f + __expf(-ot));
  if (lane == 0) m_out[(size_t)b * NH_ + h] = mn;

  // state update + h, 4 elements per lane (H=128)
  const size_t base = ((size_t)b * NH_ + h) * H_;
  const int e = lane * 4;
  v4f c4 = *(const v4f*)(c + base + e);
  v4f n4 = *(const v4f*)(n + base + e);
  v4bf z4 = *(const v4bf*)(zbf + rowx + h * H_ + e);
  v4f cn, nn;
  float hv[4], lsum = 0.f, lsq = 0.f;
#pragma unroll
  for (int q = 0; q < 4; ++q) {
    float zq = (float)z4[q];
    float cq = fg * c4[q] + ig * zq;
    float nq = fg * n4[q] + ig;
    cn[q] = cq; nn[q] = nq;
    float hq = og * (cq / (nq + EPS_));
    hv[q] = hq; lsum += hq; lsq += hq * hq;
  }
  *(v4f*)(c_out + base + e) = cn;
  *(v4f*)(n_out + base + e) = nn;

#pragma unroll
  for (int off = 16; off > 0; off >>= 1) {
    lsum += __shfl_xor(lsum, off, 32);
    lsq  += __shfl_xor(lsq, off, 32);
  }
  const float mu  = lsum * (1.f / H_);
  const float var = lsq * (1.f / H_) - mu * mu;
  const float inv = rsqrtf(var + EPS_);

  v4bf o4;
#pragma unroll
  for (int q = 0; q < 4; ++q) {
    int col = h * H_ + e + q;
    o4[q] = (bf16)((hv[q] - mu) * inv * gnw[col] + gnb[col]);
  }
  *(v4bf*)(hn + rowx + h * H_ + e) = o4;
}

// ---------------------------------------------------------------------------
// Launch
// ---------------------------------------------------------------------------
extern "C" void kernel_launch(void* const* d_in, const int* in_sizes, int n_in,
                              void* d_out, int out_size, void* d_ws, size_t ws_size,
                              hipStream_t stream) {
  const float* inputs = (const float*)d_in[0];
  const float* c      = (const float*)d_in[1];
  const float* n      = (const float*)d_in[2];
  const float* m      = (const float*)d_in[3];
  const float* conv_w = (const float*)d_in[4];
  const float* conv_b = (const float*)d_in[5];
  const float* z_w    = (const float*)d_in[6];
  const float* i_w    = (const float*)d_in[7];
  const float* i_b    = (const float*)d_in[8];
  const float* f_w    = (const float*)d_in[9];
  const float* f_b    = (const float*)d_in[10];
  const float* o_w    = (const float*)d_in[11];
  const float* o_b    = (const float*)d_in[12];
  const float* gn_w   = (const float*)d_in[13];
  const float* gn_b   = (const float*)d_in[14];
  const float* out_w  = (const float*)d_in[15];

  float* out   = (float*)d_out;                 // [B, D]
  float* c_out = out + (size_t)B_ * D_;         // [B, NH, H]
  float* n_out = c_out + (size_t)B_ * D_;       // [B, NH, H]
  float* m_out = n_out + (size_t)B_ * D_;       // [B, NH]

  char* ws = (char*)d_ws;
  bf16* Xbf = (bf16*)(ws);                        // 64 MB  inputs bf16
  bf16* Xc  = (bf16*)(ws + ((size_t)64  << 20));  // 64 MB  x_conv bf16
  bf16* Zb  = (bf16*)(ws + ((size_t)128 << 20));  // 64 MB  z bf16
  bf16* Hn  = (bf16*)(ws + ((size_t)192 << 20));  // 64 MB  h_norm bf16
  bf16* Wc  = (bf16*)(ws + ((size_t)256 << 20));  // 2 MB   conv last tap bf16
  bf16* Zw  = (bf16*)(ws + ((size_t)258 << 20));  // 2 MB   z_w bf16
  bf16* Ow  = (bf16*)(ws + ((size_t)260 << 20));  // 2 MB   out_w bf16

  {
    int n4 = (B_ * D_) / 4;
    cvt_f32_to_bf16_vec4<<<n4 / 256, 256, 0, stream>>>(inputs, Xbf, n4);
    int w4 = (D_ * D_) / 4;
    cvt_f32_to_bf16_vec4<<<w4 / 256, 256, 0, stream>>>(z_w, Zw, w4);
    cvt_f32_to_bf16_vec4<<<w4 / 256, 256, 0, stream>>>(out_w, Ow, w4);
    cvt_convw_last<<<(D_ * D_) / 256, 256, 0, stream>>>(conv_w, Wc, D_ * D_);
  }

  dim3 grid(D_ / TN, B_ / TM);
  gemm_bf16_wmma<0><<<grid, 256, 0, stream>>>(Xbf, Wc, conv_b, nullptr, Xc, D_);
  gemm_bf16_wmma<1><<<grid, 256, 0, stream>>>(Xbf, Zw, nullptr, nullptr, Zb, D_);

  gate_ln_kernel<<<(B_ * NH_) / 8, 256, 0, stream>>>(
      inputs, Xc, Zb, c, n, m, i_w, i_b, f_w, f_b, o_w, o_b, gn_w, gn_b,
      c_out, n_out, m_out, Hn);

  gemm_bf16_wmma<2><<<grid, 256, 0, stream>>>(Hn, Ow, nullptr, out, nullptr, D_);
}
